// Decoder_77180562309550
// MI455X (gfx1250) — compile-verified, matrix operand order checked by reading, not run on hardware
//
#include <hip/hip_runtime.h>

// Problem constants (from reference setup_inputs)
#define BB 2048   // batch
#define HH 256    // hidden
#define TT 32     // max length / timesteps
#define VV 128    // vocab
#define H3 768    // 3*H

typedef _Float16 half8  __attribute__((ext_vector_type(8)));
typedef _Float16 v16h   __attribute__((ext_vector_type(16)));
typedef float    v8f    __attribute__((ext_vector_type(8)));
typedef int      i4v    __attribute__((ext_vector_type(4)));

// --------------------------- CDNA5 feature probes ---------------------------
#if defined(__has_builtin)
#  if __has_builtin(__builtin_amdgcn_global_load_async_to_lds_b128) && \
      __has_builtin(__builtin_amdgcn_s_wait_asynccnt)
#    define HAS_ASYNC_LOAD 1
#  endif
#  if __has_builtin(__builtin_amdgcn_global_store_async_from_lds_b128) && \
      __has_builtin(__builtin_amdgcn_s_wait_asynccnt)
#    define HAS_ASYNC_STORE 1
#  endif
#  if __has_builtin(__builtin_amdgcn_tanhf)
#    define HAS_TANH 1
#  endif
#endif

// builtin signature (from hipcc diagnostics): void(v4i as1*, v4i as3*, Imm, Imm)
typedef __attribute__((address_space(1))) i4v* g_b128_t;
typedef __attribute__((address_space(3))) i4v* l_b128_t;

// ---------------------------------------------------------------------------
// Fragment loaders (CDNA5 WMMA 16x16x32 f16 layouts, ISA 7.12.2):
//   A (16x32, MxK): lane -> M = lane&15.  lanes<16 hold K {0..7,16..23},
//                   lanes>=16 hold K {8..15,24..31}; packed 2 halves / VGPR.
//   B (32x16, KxN): lane -> N = lane&15, same K striping.
// Both reduce to two contiguous 16-byte loads per lane.
// ---------------------------------------------------------------------------
__device__ __forceinline__ v16h load_frag(const _Float16* __restrict__ p, int hi) {
  half8 lo = *(const half8*)(p + (hi ? 8  : 0));
  half8 up = *(const half8*)(p + (hi ? 24 : 16));
  v16h r;
#pragma unroll
  for (int i = 0; i < 8; ++i) { r[i] = lo[i]; r[i + 8] = up[i]; }
  return r;
}

__device__ __forceinline__ v8f wmma_f16(v16h a, v16h b, v8f c) {
  return __builtin_amdgcn_wmma_f32_16x16x32_f16(false, a, false, b,
                                                (short)0, c, false, false);
}

// Fast activations: raw v_rcp_f32 / v_exp_f32 / v_tanh_f32 (no IEEE div chain)
__device__ __forceinline__ float fast_rcp(float x) { return __builtin_amdgcn_rcpf(x); }
__device__ __forceinline__ float sigmoid_fast(float x) {
  return fast_rcp(1.0f + __expf(-x));
}
__device__ __forceinline__ float tanh_fast(float x) {
#if defined(HAS_TANH)
  return __builtin_amdgcn_tanhf(x);
#else
  // tanh(x) = 1 - 2/(exp(2x)+1); saturates correctly at +/-1 for large |x|
  return 1.0f - 2.0f * fast_rcp(__expf(2.0f * x) + 1.0f);
#endif
}

// ---------------------------------------------------------------------------
// f32 -> f16 conversion (weights)
// ---------------------------------------------------------------------------
__global__ void cvt_f32_to_f16(const float* __restrict__ in,
                               _Float16* __restrict__ out, int n) {
  int i = blockIdx.x * 256 + threadIdx.x;
  if (i < n) out[i] = (_Float16)in[i];
}

// ---------------------------------------------------------------------------
// Embedding + ReLU -> f16, time-major [T*B, H]
// ---------------------------------------------------------------------------
__global__ __launch_bounds__(HH)
void embed_relu_kernel(const int* __restrict__ tgt,      // [B, T] int32
                       const float* __restrict__ emb,    // [V, H] f32
                       _Float16* __restrict__ x0)        // [T*B, H] f16
{
  int row = blockIdx.x;              // row = t*B + b
  int t = row >> 11;                 // / 2048
  int b = row & (BB - 1);
  int tok = (t == 0) ? 0 : tgt[b * TT + (t - 1)];
  float v = emb[tok * HH + threadIdx.x];
  x0[(size_t)row * HH + threadIdx.x] = (_Float16)fmaxf(v, 0.0f);
}

// ---------------------------------------------------------------------------
// One GRU layer, all T steps.
// Block = 32 batch rows (two 16-row M-tiles), 16 waves (512 thr).
// Wave w owns H-column tile w (16 cols) and computes r/z/n gate columns for
// BOTH M-tiles, so each weight B-fragment feeds two WMMAs.
// x_t (16 KB, contiguous) staged into LDS via async global->LDS when
// available; y_t written back linearly from the updated LDS h-tile.
// ---------------------------------------------------------------------------
__global__ __launch_bounds__(512)
void gru_layer_kernel(const _Float16* __restrict__ x_in,   // [T*B, H] f16
                      _Float16* __restrict__ x_out,        // [T*B, H] f16
                      const _Float16* __restrict__ Wih,    // [3H, H] f16
                      const _Float16* __restrict__ Whh,    // [3H, H] f16
                      const float* __restrict__ bih,       // [3H]
                      const float* __restrict__ bhh,       // [3H]
                      const float* __restrict__ h0,        // [B, H] f32
                      float* __restrict__ h_last)          // [B, H] f32
{
  __shared__ _Float16 hsh[32 * HH];                        // 16 KB
  __shared__ _Float16 xsh[32 * HH];                        // 16 KB

  const int tid  = threadIdx.x;
  const int wave = tid >> 5;          // 0..15  -> H-tile, col base = wave*16
  const int lane = tid & 31;
  const int ln   = lane & 15;
  const int hi   = lane >> 4;         // 0/1
  const int m0   = blockIdx.x * 32;   // batch-row base
  const int col  = wave * 16 + ln;    // owned hidden column

  float hreg[2][8];                   // f32 hidden, C-tile layout, per M-tile

  // init hidden from h0 (f32) and stage f16 copy into LDS
#pragma unroll
  for (int mt = 0; mt < 2; ++mt) {
#pragma unroll
    for (int r = 0; r < 8; ++r) {
      int lrow = mt * 16 + r + hi * 8;
      float v = h0[(size_t)(m0 + lrow) * HH + col];
      hreg[mt][r] = v;
      hsh[lrow * HH + col] = (_Float16)v;
    }
  }
  __syncthreads();

  // per-column gate biases (loop-invariant over t)
  float bx_g[3], bh_g[3];
#pragma unroll
  for (int g = 0; g < 3; ++g) { bx_g[g] = bih[g * HH + col]; bh_g[g] = bhh[g * HH + col]; }

#pragma unroll 1
  for (int t = 0; t < TT; ++t) {
    // ---- stage x_t tile (32 rows x 256 cols f16 = 16 KB, contiguous) ----
    {
      const _Float16* src = x_in + (size_t)(t * BB + m0) * HH;
#if defined(HAS_ASYNC_LOAD)
      __builtin_amdgcn_global_load_async_to_lds_b128(
          (g_b128_t)((char*)src + tid * 32),
          (l_b128_t)((char*)xsh + tid * 32), 0, 0);
      __builtin_amdgcn_global_load_async_to_lds_b128(
          (g_b128_t)((char*)src + tid * 32 + 16),
          (l_b128_t)((char*)xsh + tid * 32 + 16), 0, 0);
      __builtin_amdgcn_s_wait_asynccnt(0);
#else
      uint4 a = *(const uint4*)((const char*)src + tid * 32);
      uint4 b = *(const uint4*)((const char*)src + tid * 32 + 16);
      *(uint4*)((char*)xsh + tid * 32)      = a;
      *(uint4*)((char*)xsh + tid * 32 + 16) = b;
#endif
    }
    __syncthreads();

    // ---- dual-projection GEMM: acc = [x_t;h] @ [Wih;Whh]^T + bias ----
    v8f accx[2][3], acch[2][3];
#pragma unroll
    for (int g = 0; g < 3; ++g)
#pragma unroll
      for (int mt = 0; mt < 2; ++mt)
#pragma unroll
        for (int r = 0; r < 8; ++r) { accx[mt][g][r] = bx_g[g]; acch[mt][g][r] = bh_g[g]; }

    const _Float16* xr0 = &xsh[ln * HH];
    const _Float16* xr1 = &xsh[(16 + ln) * HH];
    const _Float16* hr0 = &hsh[ln * HH];
    const _Float16* hr1 = &hsh[(16 + ln) * HH];

#pragma unroll 1
    for (int kt = 0; kt < 8; ++kt) {
      int kb = kt * 32;
      v16h Ax0 = load_frag(xr0 + kb, hi);
      v16h Ax1 = load_frag(xr1 + kb, hi);
      v16h Ah0 = load_frag(hr0 + kb, hi);
      v16h Ah1 = load_frag(hr1 + kb, hi);
#pragma unroll
      for (int g = 0; g < 3; ++g) {
        // B-fragment row = output column index within [3H]
        v16h Bx = load_frag(Wih + (size_t)(g * HH + col) * HH + kb, hi);
        accx[0][g] = wmma_f16(Ax0, Bx, accx[0][g]);
        accx[1][g] = wmma_f16(Ax1, Bx, accx[1][g]);
        v16h Bh = load_frag(Whh + (size_t)(g * HH + col) * HH + kb, hi);
        acch[0][g] = wmma_f16(Ah0, Bh, acch[0][g]);
        acch[1][g] = wmma_f16(Ah1, Bh, acch[1][g]);
      }
    }

    // ---- gates + state update (registers only) ----
    _Float16 hnew16[2][8];
#pragma unroll
    for (int mt = 0; mt < 2; ++mt) {
#pragma unroll
      for (int r = 0; r < 8; ++r) {
        float rg = sigmoid_fast(accx[mt][0][r] + acch[mt][0][r]);
        float zg = sigmoid_fast(accx[mt][1][r] + acch[mt][1][r]);
        float ng = tanh_fast(accx[mt][2][r] + rg * acch[mt][2][r]);
        float hn = (1.0f - zg) * ng + zg * hreg[mt][r];
        hreg[mt][r] = hn;
        hnew16[mt][r] = (_Float16)hn;
      }
    }

    __syncthreads();   // all waves done reading old h / x
#pragma unroll
    for (int mt = 0; mt < 2; ++mt)
#pragma unroll
      for (int r = 0; r < 8; ++r)
        hsh[(mt * 16 + r + hi * 8) * HH + col] = hnew16[mt][r];
    __syncthreads();   // hsh now holds y_t = h_new for the whole tile

    // ---- linear y_t writeback from LDS (hsh layout == x_out tile layout) ----
    {
      _Float16* dst = x_out + (size_t)(t * BB + m0) * HH;
#if defined(HAS_ASYNC_STORE)
      __builtin_amdgcn_global_store_async_from_lds_b128(
          (g_b128_t)((char*)dst + tid * 32),
          (l_b128_t)((char*)hsh + tid * 32), 0, 0);
      __builtin_amdgcn_global_store_async_from_lds_b128(
          (g_b128_t)((char*)dst + tid * 32 + 16),
          (l_b128_t)((char*)hsh + tid * 32 + 16), 0, 0);
      // completion enforced by next iteration's s_wait_asynccnt / s_endpgm
#else
      uint4 a = *(const uint4*)((const char*)hsh + tid * 32);
      uint4 b = *(const uint4*)((const char*)hsh + tid * 32 + 16);
      *(uint4*)((char*)dst + tid * 32)      = a;
      *(uint4*)((char*)dst + tid * 32 + 16) = b;
#endif
    }
#if defined(HAS_ASYNC_STORE) && !defined(HAS_ASYNC_LOAD)
    __builtin_amdgcn_s_wait_asynccnt(0);  // protect hsh before next overwrite
#endif
  }

  // final hidden (f32) -> d_out hidden section
#pragma unroll
  for (int mt = 0; mt < 2; ++mt)
#pragma unroll
    for (int r = 0; r < 8; ++r)
      h_last[(size_t)(m0 + mt * 16 + r + hi * 8) * HH + col] = hreg[mt][r];
}

// ---------------------------------------------------------------------------
// logits = y @ fc_w^T + fc_b ; log_softmax over V=128. One wave / 16 rows.
// For fixed accumulator VGPR r, lanes 0..15 (resp 16..31) hold the SAME row,
// different columns -> shfl_xor reduction within each 16-lane half.
// ---------------------------------------------------------------------------
__global__ __launch_bounds__(32)
void fc_logsoftmax_kernel(const _Float16* __restrict__ x,    // [T*B, H] f16
                          const _Float16* __restrict__ fcw,  // [V, H]  f16
                          const float* __restrict__ fcb,     // [V]
                          float* __restrict__ out)           // [B, T, V]
{
  const int lane = threadIdx.x;
  const int ln = lane & 15;
  const int hi = lane >> 4;
  const int m0 = blockIdx.x * 16;

  v8f acc[8];
#pragma unroll
  for (int nt = 0; nt < 8; ++nt) {
    float b = fcb[nt * 16 + ln];
#pragma unroll
    for (int r = 0; r < 8; ++r) acc[nt][r] = b;
  }

  const _Float16* xrow = x + (size_t)(m0 + ln) * HH;
#pragma unroll 1
  for (int kt = 0; kt < 8; ++kt) {
    v16h A = load_frag(xrow + kt * 32, hi);
#pragma unroll
    for (int nt = 0; nt < 8; ++nt) {
      v16h Bf = load_frag(fcw + (size_t)(nt * 16 + ln) * HH + kt * 32, hi);
      acc[nt] = wmma_f16(A, Bf, acc[nt]);
    }
  }

#pragma unroll
  for (int r = 0; r < 8; ++r) {
    float mx = acc[0][r];
#pragma unroll
    for (int nt = 1; nt < 8; ++nt) mx = fmaxf(mx, acc[nt][r]);
#pragma unroll
    for (int s = 1; s < 16; s <<= 1) mx = fmaxf(mx, __shfl_xor(mx, s, 32));
    float sum = 0.0f;
#pragma unroll
    for (int nt = 0; nt < 8; ++nt) sum += __expf(acc[nt][r] - mx);
#pragma unroll
    for (int s = 1; s < 16; s <<= 1) sum += __shfl_xor(sum, s, 32);
    float lse = mx + __logf(sum);

    int m = m0 + r + hi * 8;      // global row = t*B + b
    int t = m >> 11;
    int b = m & (BB - 1);
    float* orow = out + ((size_t)b * TT + t) * VV;
#pragma unroll
    for (int nt = 0; nt < 8; ++nt) orow[nt * 16 + ln] = acc[nt][r] - lse;
  }
}

// ---------------------------------------------------------------------------
// Host-side launch
// Inputs (setup_inputs order):
//  0 enc_outputs (unused)  1 enc_hidden [L,B,H] f32  2 tgt [B,T] int
//  3 embed_w [V,H] f32     4 W_ih [L,3H,H]           5 W_hh [L,3H,H]
//  6 b_ih [L,3H]           7 b_hh [L,3H]             8 fc_w [V,H]  9 fc_b [V]
// d_out: log_probs [B,T,V] f32 then hidden [L,B,H] f32
// ---------------------------------------------------------------------------
extern "C" void kernel_launch(void* const* d_in, const int* in_sizes, int n_in,
                              void* d_out, int out_size, void* d_ws, size_t ws_size,
                              hipStream_t stream) {
  const float* enc_hidden = (const float*)d_in[1];
  const int*   tgt        = (const int*)d_in[2];
  const float* embed_w    = (const float*)d_in[3];
  const float* W_ih       = (const float*)d_in[4];
  const float* W_hh       = (const float*)d_in[5];
  const float* b_ih       = (const float*)d_in[6];
  const float* b_hh       = (const float*)d_in[7];
  const float* fc_w       = (const float*)d_in[8];
  const float* fc_b       = (const float*)d_in[9];
  float* out = (float*)d_out;

  // workspace carve-up (256B aligned)
  char* ws = (char*)d_ws;
  size_t off = 0;
  auto carve = [&](size_t bytes) -> void* {
    void* p = ws + off;
    off = (off + bytes + 255) & ~(size_t)255;
    return p;
  };
  const size_t xbytes = (size_t)TT * BB * HH * sizeof(_Float16);   // 33.5 MB
  _Float16* bufA   = (_Float16*)carve(xbytes);   // x0 (embed) / layer-1 output
  _Float16* bufB   = (_Float16*)carve(xbytes);   // layer-0 output
  _Float16* Wih16  = (_Float16*)carve((size_t)2 * H3 * HH * sizeof(_Float16));
  _Float16* Whh16  = (_Float16*)carve((size_t)2 * H3 * HH * sizeof(_Float16));
  _Float16* fcw16  = (_Float16*)carve((size_t)VV * HH * sizeof(_Float16));

  // 1) weight conversion to f16
  {
    int n = 2 * H3 * HH;
    cvt_f32_to_f16<<<(n + 255) / 256, 256, 0, stream>>>(W_ih, Wih16, n);
    cvt_f32_to_f16<<<(n + 255) / 256, 256, 0, stream>>>(W_hh, Whh16, n);
    int nf = VV * HH;
    cvt_f32_to_f16<<<(nf + 255) / 256, 256, 0, stream>>>(fc_w, fcw16, nf);
  }

  // 2) embedding + relu -> bufA  [T*B, H] f16
  embed_relu_kernel<<<TT * BB, HH, 0, stream>>>(tgt, embed_w, bufA);

  const size_t hidden_out_off = (size_t)BB * TT * VV;

  // 3) GRU layer 0: bufA -> bufB
  gru_layer_kernel<<<BB / 32, 512, 0, stream>>>(
      bufA, bufB,
      Wih16, Whh16,
      b_ih, b_hh,
      enc_hidden,
      out + hidden_out_off);

  // 4) GRU layer 1: bufB -> bufA
  gru_layer_kernel<<<BB / 32, 512, 0, stream>>>(
      bufB, bufA,
      Wih16 + (size_t)H3 * HH, Whh16 + (size_t)H3 * HH,
      b_ih + H3, b_hh + H3,
      enc_hidden + (size_t)BB * HH,
      out + hidden_out_off + (size_t)BB * HH);

  // 5) FC + log_softmax: bufA -> out [B,T,V]
  fc_logsoftmax_kernel<<<(TT * BB) / 16, 32, 0, stream>>>(bufA, fcw16, fc_b, out);
}
